// AttentionEncoder_18717467476193
// MI455X (gfx1250) — matrix operand order repaired and checked by reading, not compiled, not run on hardware
//
#include <hip/hip_runtime.h>
#include <hip/hip_bf16.h>

#define DEVI static __device__ __forceinline__

typedef __bf16 bf16;
typedef __attribute__((ext_vector_type(2)))  __bf16 v2bf;
typedef __attribute__((ext_vector_type(4)))  __bf16 v4bf;
typedef __attribute__((ext_vector_type(8)))  __bf16 v8bf;
typedef __attribute__((ext_vector_type(16))) __bf16 v16bf;
typedef __attribute__((ext_vector_type(8)))  float  v8f;

namespace cfg {
constexpr int B = 8, S = 2048, D = 1024, K = 64;
constexpr int BS = B * S;                          // 16384 rows
constexpr float SCALE = 0.125f;                    // 1/sqrt(64)
constexpr float EPS = 1e-5f;
// workspace byte offsets
constexpr size_t XB   = 0;                         // x bf16        [BS][D]
constexpr size_t WKT  = XB   + (size_t)BS*D*2;     // Wk^T bf16     [K][D] (pre-scaled)
constexpr size_t WQT  = WKT  + (size_t)K*D*2;      // Wq^T bf16     [K][D]
constexpr size_t WVT  = WQT  + (size_t)K*D*2;      // Wv^T bf16     [D][D]
constexpr size_t W1T  = WVT  + (size_t)D*D*2;      // W1^T bf16     [D][D]
constexpr size_t KS   = W1T  + (size_t)D*D*2;      // keys*scale    [BS][K] bf16
constexpr size_t QS   = KS   + (size_t)BS*K*2;     // queries       [BS][K] bf16
constexpr size_t VS   = QS   + (size_t)BS*K*2;     // values        [BS][D] bf16
constexpr size_t BIAS = VS   + (size_t)BS*D*2;     // biases f32    [BS]
constexpr size_t RESB = BIAS + (size_t)BS*4;       // res bf16      [BS][D]
constexpr size_t TMP  = RESB + (size_t)BS*D*2;     // f32 scratch   [BS][D]
}

// ---------------- CDNA5 async memory->LDS DMA --------------------------------
// GLOBAL_LOAD_ASYNC_TO_LDS_B128: per-lane 16B copy global->LDS, ASYNCcnt
// tracked, no VGPR staging. lds_off = byte offset within LDS (low 32 bits of
// the generic shared-space pointer).

DEVI void async_copy_b128(unsigned lds_off, const void* gaddr) {
  asm volatile("global_load_async_to_lds_b128 %0, %1, off"
               :: "v"(lds_off), "v"(gaddr) : "memory");
}
DEVI void wait_async() { asm volatile("s_wait_asynccnt 0x0" ::: "memory"); }

// ---------------- WMMA helpers (CDNA5 16x16x32 bf16, f32 accum) --------------

DEVI v16bf cat8(v8bf lo, v8bf hi) {
  return __builtin_shufflevector(lo, hi, 0,1,2,3,4,5,6,7,8,9,10,11,12,13,14,15);
}

// A operand: 16x32 bf16 tile, row-major LDS tile with stride lda (elements).
// lanes 0-15: rows M=0..15, K = 0..7 and 16..23 ; lanes 16-31: K = 8..15 and 24..31
DEVI v16bf load_a_frag(const bf16* tile, int lda, int mbase, int kb, int lane) {
  const bf16* p = tile + (size_t)(mbase + (lane & 15)) * lda + kb + ((lane >> 4) & 1) * 8;
  return cat8(*(const v8bf*)p, *(const v8bf*)(p + 16));
}

// B operand: 32x16 bf16 (KxN), tile stored N-major [n][k] with stride ldb.
// lanes 0-15: col N=lane, K = 0..7 and 16..23 ; lanes 16-31: col N=lane-16, K = 8..15 and 24..31
DEVI v16bf load_b_frag(const bf16* tile, int ldb, int nbase, int kb, int lane) {
  const bf16* p = tile + (size_t)(nbase + (lane & 15)) * ldb + kb + ((lane >> 4) & 1) * 8;
  return cat8(*(const v8bf*)p, *(const v8bf*)(p + 16));
}

DEVI v8f wmma_bf16(v16bf a, v16bf b, v8f c) {
  return __builtin_amdgcn_wmma_f32_16x16x32_bf16(false, a, false, b, (short)0, c, false, false);
}

DEVI v8f vzero8() {
  v8f z;
  #pragma unroll
  for (int i = 0; i < 8; ++i) z[i] = 0.f;
  return z;
}

// ---------------- small prep kernels ----------------------------------------

__global__ __launch_bounds__(256)
void cvt_f32_bf16_kernel(const float* __restrict__ src, bf16* __restrict__ dst, long long n) {
  long long i = ((long long)blockIdx.x * blockDim.x + threadIdx.x) * 8;
  if (i + 8 > n) return;
  float4 a = *(const float4*)(src + i);
  float4 b = *(const float4*)(src + i + 4);
  v8bf o;
  o[0]=(bf16)a.x; o[1]=(bf16)a.y; o[2]=(bf16)a.z; o[3]=(bf16)a.w;
  o[4]=(bf16)b.x; o[5]=(bf16)b.y; o[6]=(bf16)b.z; o[7]=(bf16)b.w;
  *(v8bf*)(dst + i) = o;
}

// W f32 [D][N] -> Wt bf16 [N][D], scaled. grid=(N/32, D/32), block=256.
__global__ __launch_bounds__(256)
void wtrans_kernel(const float* __restrict__ W, bf16* __restrict__ Wt, int N, float scale) {
  __shared__ float tile[32][33];
  int nt = blockIdx.x * 32, kt = blockIdx.y * 32;
  int t = threadIdx.x, c = t & 31, r0 = t >> 5;
  #pragma unroll
  for (int i = 0; i < 4; ++i) {
    int r = r0 + i * 8;
    tile[r][c] = W[(size_t)(kt + r) * N + (nt + c)] * scale;
  }
  __syncthreads();
  #pragma unroll
  for (int i = 0; i < 4; ++i) {
    int r = r0 + i * 8;
    Wt[(size_t)(nt + r) * cfg::D + (kt + c)] = (bf16)tile[c][r];
  }
}

// biases[r] = dot(x[r,:], Wb[:,0]) + bb ; one wave per row, 8 rows per block.
__global__ __launch_bounds__(256)
void biasproj_kernel(const float* __restrict__ x, const float* __restrict__ Wb,
                     const float* __restrict__ bb, float* __restrict__ outb) {
  int wid = threadIdx.x >> 5, lane = threadIdx.x & 31;
  int row = blockIdx.x * 8 + wid;
  const float* xr = x + (size_t)row * cfg::D;
  float s = 0.f;
  for (int e = lane; e < cfg::D; e += 32) s += xr[e] * Wb[e];
  #pragma unroll
  for (int m = 16; m >= 1; m >>= 1) s += __shfl_xor(s, m, 32);
  if (lane == 0) outb[row] = s + bb[0];
}

// ---------------- generic WMMA GEMM: C[M,N] = A[M,1024] @ Bt[N,1024]^T -------
// tile 128x128, 256 threads = 8 waves (4 M-rows x 2 N-cols of 32x64 each).
// epilogue: v = acc + alpha*bias[n]; optional relu; write bf16 or f32 (stride N)

__global__ __launch_bounds__(256)
void gemm_kernel(const bf16* __restrict__ A, const bf16* __restrict__ Bt,
                 const float* __restrict__ bias, float alpha, int relu, int N,
                 bf16* __restrict__ outb, float* __restrict__ outf) {
  __shared__ bf16 As[128 * 72];
  __shared__ bf16 Bs[128 * 72];
  const int t = threadIdx.x, lane = t & 31, wid = t >> 5;
  const int wm = wid & 3, wn = wid >> 2;
  const int mtile = blockIdx.y * 128, ntile = blockIdx.x * 128;

  v8f acc[2][4];
  #pragma unroll
  for (int mi = 0; mi < 2; ++mi)
    #pragma unroll
    for (int ni = 0; ni < 4; ++ni) acc[mi][ni] = vzero8();

  const int lrow = t >> 1, lseg = (t & 1) * 32;
  for (int kc = 0; kc < cfg::D; kc += 64) {
    { // stage A tile 128x64 via async DMA to LDS
      const bf16* g = A + (size_t)(mtile + lrow) * cfg::D + kc + lseg;
      unsigned l = (unsigned)(size_t)&As[lrow * 72 + lseg];
      #pragma unroll
      for (int q = 0; q < 4; ++q) async_copy_b128(l + q * 16, g + q * 8);
      if (kc + 64 < cfg::D) __builtin_prefetch(g + 64, 0, 1);
    }
    { // stage B tile 128x64 (rows are output columns), zero past N
      int brow = ntile + lrow;
      unsigned l = (unsigned)(size_t)&Bs[lrow * 72 + lseg];
      if (brow < N) {
        const bf16* g = Bt + (size_t)brow * cfg::D + kc + lseg;
        #pragma unroll
        for (int q = 0; q < 4; ++q) async_copy_b128(l + q * 16, g + q * 8);
      } else {
        uint4 z = make_uint4(0, 0, 0, 0);
        #pragma unroll
        for (int q = 0; q < 4; ++q) *(uint4*)&Bs[lrow * 72 + lseg + q * 8] = z;
      }
    }
    wait_async();
    __syncthreads();
    #pragma unroll
    for (int kb = 0; kb < 64; kb += 32) {
      v16bf af[2], bfv[4];
      af[0] = load_a_frag(As, 72, wm * 32,      kb, lane);
      af[1] = load_a_frag(As, 72, wm * 32 + 16, kb, lane);
      #pragma unroll
      for (int ni = 0; ni < 4; ++ni)
        bfv[ni] = load_b_frag(Bs, 72, wn * 64 + ni * 16, kb, lane);
      #pragma unroll
      for (int mi = 0; mi < 2; ++mi)
        #pragma unroll
        for (int ni = 0; ni < 4; ++ni)
          acc[mi][ni] = wmma_bf16(af[mi], bfv[ni], acc[mi][ni]);
    }
    __syncthreads();
  }

  const int half = (lane >> 4) & 1, c0 = lane & 15;
  #pragma unroll
  for (int mi = 0; mi < 2; ++mi)
    #pragma unroll
    for (int ni = 0; ni < 4; ++ni)
      #pragma unroll
      for (int g = 0; g < 8; ++g) {
        int gn = ntile + wn * 64 + ni * 16 + c0;
        if (gn >= N) continue;
        int gm = mtile + wm * 32 + mi * 16 + half * 8 + g;
        float v = acc[mi][ni][g] + alpha * bias[gn];
        if (relu) v = fmaxf(v, 0.f);
        if (outb) outb[(size_t)gm * N + gn] = (bf16)v;
        else      outf[(size_t)gm * N + gn] = v;
      }
}

// ---------------- fused sigmoid-attention ------------------------------------
// out[b, i0:i0+128, d0:d0+128] = sum_j sigmoid(Kc[i]·Q[j] + bias[i]) * V[j, d]
// grid = (D/128, S/128, B), block = 256 (8 waves)

__global__ __launch_bounds__(256)
void attn_kernel(const bf16* __restrict__ Kc, const bf16* __restrict__ Q,
                 const bf16* __restrict__ V, const float* __restrict__ biasv,
                 float* __restrict__ out) {
  __shared__ bf16 Ak[128 * 72];     // keys*scale tile   [i][k]
  __shared__ bf16 Bq[128 * 72];     // queries tile      [j][k] (B-operand layout)
  __shared__ bf16 Sc[128 * 136];    // sigmoid scores    [i][j] (A-operand layout)
  __shared__ bf16 Vt[128 * 136];    // values transposed [d][j] (B-operand layout)
  __shared__ float sb[128];

  const int t = threadIdx.x, lane = t & 31, wid = t >> 5;
  const int wm = wid & 3, wn = wid >> 2;
  const int d0 = blockIdx.x * 128, i0 = blockIdx.y * 128, b = blockIdx.z;
  const size_t rowbase = (size_t)b * cfg::S + i0;
  const int lrow = t >> 1, lseg = (t & 1) * 32;

  { // keys tile (once, async) + row biases
    const bf16* g = Kc + (rowbase + lrow) * cfg::K + lseg;
    unsigned l = (unsigned)(size_t)&Ak[lrow * 72 + lseg];
    #pragma unroll
    for (int q = 0; q < 2; ++q) async_copy_b128(l + q * 16, g + q * 8);
    if (t < 128) sb[t] = biasv[rowbase + t];
  }

  v8f oacc[2][4];
  #pragma unroll
  for (int mi = 0; mi < 2; ++mi)
    #pragma unroll
    for (int ni = 0; ni < 4; ++ni) oacc[mi][ni] = vzero8();

  for (int jt = 0; jt < cfg::S; jt += 128) {
    __syncthreads();
    { // queries tile [j][0:64] via async DMA
      const bf16* g = Q + ((size_t)b * cfg::S + jt + lrow) * cfg::K + lseg;
      unsigned l = (unsigned)(size_t)&Bq[lrow * 72 + lseg];
      #pragma unroll
      for (int q = 0; q < 2; ++q) async_copy_b128(l + q * 16, g + q * 8);
    }
    { // V tile transposed into Vt[d][j]; thread: 2 j-rows x 32 d-cols
      int jp = t & 63, dg = t >> 6;
      int jj = jp * 2;
      const bf16* r0p = V + ((size_t)b * cfg::S + jt + jj) * cfg::D + d0 + dg * 32;
      const bf16* r1p = r0p + cfg::D;
      v8bf r0[4], r1[4];
      #pragma unroll
      for (int q = 0; q < 4; ++q) {
        r0[q] = *(const v8bf*)(r0p + q * 8);
        r1[q] = *(const v8bf*)(r1p + q * 8);
      }
      #pragma unroll
      for (int e = 0; e < 32; ++e) {
        v2bf pr; pr[0] = r0[e >> 3][e & 7]; pr[1] = r1[e >> 3][e & 7];
        *(v2bf*)&Vt[(dg * 32 + e) * 136 + jj] = pr;
      }
    }
    wait_async();
    __syncthreads();

    { // scores tile: 128x128 = Kc_tile @ Q_tile^T over K=64
      v8f sacc[2][4];
      #pragma unroll
      for (int mi = 0; mi < 2; ++mi)
        #pragma unroll
        for (int ni = 0; ni < 4; ++ni) sacc[mi][ni] = vzero8();
      #pragma unroll
      for (int kb = 0; kb < 64; kb += 32) {
        v16bf af[2], bfv[4];
        af[0] = load_a_frag(Ak, 72, wm * 32,      kb, lane);
        af[1] = load_a_frag(Ak, 72, wm * 32 + 16, kb, lane);
        #pragma unroll
        for (int ni = 0; ni < 4; ++ni)
          bfv[ni] = load_b_frag(Bq, 72, wn * 64 + ni * 16, kb, lane);
        #pragma unroll
        for (int mi = 0; mi < 2; ++mi)
          #pragma unroll
          for (int ni = 0; ni < 4; ++ni)
            sacc[mi][ni] = wmma_bf16(af[mi], bfv[ni], sacc[mi][ni]);
      }
      const int half = (lane >> 4) & 1, c0 = lane & 15;
      #pragma unroll
      for (int mi = 0; mi < 2; ++mi)
        #pragma unroll
        for (int ni = 0; ni < 4; ++ni)
          #pragma unroll
          for (int g = 0; g < 8; ++g) {
            int r = wm * 32 + mi * 16 + half * 8 + g;
            int c = wn * 64 + ni * 16 + c0;
            float v = sacc[mi][ni][g] + sb[r];
            v = 1.f / (1.f + __expf(-v));
            Sc[r * 136 + c] = (bf16)v;
          }
    }
    __syncthreads();

    // out tile += Sc(128x128) @ Vt^T  (K dim = j, 4 steps of 32)
    #pragma unroll
    for (int kb = 0; kb < 128; kb += 32) {
      v16bf af[2], bfv[4];
      af[0] = load_a_frag(Sc, 136, wm * 32,      kb, lane);
      af[1] = load_a_frag(Sc, 136, wm * 32 + 16, kb, lane);
      #pragma unroll
      for (int ni = 0; ni < 4; ++ni)
        bfv[ni] = load_b_frag(Vt, 136, wn * 64 + ni * 16, kb, lane);
      #pragma unroll
      for (int mi = 0; mi < 2; ++mi)
        #pragma unroll
        for (int ni = 0; ni < 4; ++ni)
          oacc[mi][ni] = wmma_bf16(af[mi], bfv[ni], oacc[mi][ni]);
    }
  }

  const int half = (lane >> 4) & 1, c0 = lane & 15;
  #pragma unroll
  for (int mi = 0; mi < 2; ++mi)
    #pragma unroll
    for (int ni = 0; ni < 4; ++ni)
      #pragma unroll
      for (int g = 0; g < 8; ++g) {
        size_t r = rowbase + wm * 32 + mi * 16 + half * 8 + g;
        int c = d0 + wn * 64 + ni * 16 + c0;
        out[r * cfg::D + c] = oacc[mi][ni][g];
      }
}

// ---------------- layernorm kernels ------------------------------------------

DEVI float2 block_reduce2(float a, float b, float* sbuf, int t) {
  #pragma unroll
  for (int m = 16; m >= 1; m >>= 1) {
    a += __shfl_xor(a, m, 32);
    b += __shfl_xor(b, m, 32);
  }
  int lane = t & 31, wid = t >> 5;
  if (lane == 0) { sbuf[wid] = a; sbuf[8 + wid] = b; }
  __syncthreads();
  if (t == 0) {
    float ta = 0.f, tb = 0.f;
    for (int w = 0; w < 8; ++w) { ta += sbuf[w]; tb += sbuf[8 + w]; }
    sbuf[0] = ta; sbuf[8] = tb;
  }
  __syncthreads();
  return make_float2(sbuf[0], sbuf[8]);
}

// res = LN(x + attn) * g1 + be1  -> bf16
__global__ __launch_bounds__(256)
void ln1_kernel(const float* __restrict__ x, const float* __restrict__ attn,
                const float* __restrict__ g1, const float* __restrict__ be1,
                bf16* __restrict__ res) {
  __shared__ float sred[16];
  const int t = threadIdx.x;
  const size_t row = blockIdx.x;
  float4 xv = *(const float4*)(x + row * cfg::D + t * 4);
  float4 av = *(const float4*)(attn + row * cfg::D + t * 4);
  float v[4] = {xv.x + av.x, xv.y + av.y, xv.z + av.z, xv.w + av.w};
  float s1 = v[0] + v[1] + v[2] + v[3];
  float s2 = v[0]*v[0] + v[1]*v[1] + v[2]*v[2] + v[3]*v[3];
  float2 s = block_reduce2(s1, s2, sred, t);
  float mean = s.x * (1.f / cfg::D);
  float var  = s.y * (1.f / cfg::D) - mean * mean;
  float rstd = rsqrtf(var + cfg::EPS);
  #pragma unroll
  for (int i = 0; i < 4; ++i) {
    int c = t * 4 + i;
    res[row * cfg::D + c] = (bf16)(((v[i] - mean) * rstd) * g1[c] + be1[c]);
  }
}

// out = LN(res + ff)  -> f32
__global__ __launch_bounds__(256)
void ln2_kernel(const bf16* __restrict__ res, const float* __restrict__ ff,
                float* __restrict__ out) {
  __shared__ float sred[16];
  const int t = threadIdx.x;
  const size_t row = blockIdx.x;
  v4bf rv = *(const v4bf*)(res + row * cfg::D + t * 4);
  float4 fv = *(const float4*)(ff + row * cfg::D + t * 4);
  float v[4] = {(float)rv[0] + fv.x, (float)rv[1] + fv.y,
                (float)rv[2] + fv.z, (float)rv[3] + fv.w};
  float s1 = v[0] + v[1] + v[2] + v[3];
  float s2 = v[0]*v[0] + v[1]*v[1] + v[2]*v[2] + v[3]*v[3];
  float2 s = block_reduce2(s1, s2, sred, t);
  float mean = s.x * (1.f / cfg::D);
  float var  = s.y * (1.f / cfg::D) - mean * mean;
  float rstd = rsqrtf(var + cfg::EPS);
  #pragma unroll
  for (int i = 0; i < 4; ++i)
    out[row * cfg::D + t * 4 + i] = (v[i] - mean) * rstd;
}

// ---------------- host orchestration -----------------------------------------

extern "C" void kernel_launch(void* const* d_in, const int* in_sizes, int n_in,
                              void* d_out, int out_size, void* d_ws, size_t ws_size,
                              hipStream_t stream) {
  using namespace cfg;
  const float* x   = (const float*)d_in[0];
  const float* Wk  = (const float*)d_in[1];
  const float* bk  = (const float*)d_in[2];
  const float* Wq  = (const float*)d_in[3];
  const float* bq  = (const float*)d_in[4];
  const float* Wv  = (const float*)d_in[5];
  const float* bv  = (const float*)d_in[6];
  const float* Wb  = (const float*)d_in[7];
  const float* bb  = (const float*)d_in[8];
  const float* W1  = (const float*)d_in[9];
  const float* b1  = (const float*)d_in[10];
  const float* g1  = (const float*)d_in[11];
  const float* be1 = (const float*)d_in[12];

  char* ws = (char*)d_ws;
  bf16*  xb    = (bf16*)(ws + XB);
  bf16*  wkt   = (bf16*)(ws + WKT);
  bf16*  wqt   = (bf16*)(ws + WQT);
  bf16*  wvt   = (bf16*)(ws + WVT);
  bf16*  w1t   = (bf16*)(ws + W1T);
  bf16*  keysc = (bf16*)(ws + KS);
  bf16*  qbf   = (bf16*)(ws + QS);
  bf16*  vbf   = (bf16*)(ws + VS);
  float* biasb = (float*)(ws + BIAS);
  bf16*  resb  = (bf16*)(ws + RESB);
  float* tmp   = (float*)(ws + TMP);
  float* outp  = (float*)d_out;

  const long long nx = (long long)BS * D;
  cvt_f32_bf16_kernel<<<(int)(nx / (256 * 8)), 256, 0, stream>>>(x, xb, nx);

  wtrans_kernel<<<dim3(K / 32, D / 32), 256, 0, stream>>>(Wk, wkt, K, SCALE);
  wtrans_kernel<<<dim3(K / 32, D / 32), 256, 0, stream>>>(Wq, wqt, K, 1.f);
  wtrans_kernel<<<dim3(D / 32, D / 32), 256, 0, stream>>>(Wv, wvt, D, 1.f);
  wtrans_kernel<<<dim3(D / 32, D / 32), 256, 0, stream>>>(W1, w1t, D, 1.f);

  biasproj_kernel<<<BS / 8, 256, 0, stream>>>(x, Wb, bb, biasb);

  // keys*scale (Wk pre-scaled; bias also scaled), queries, values
  gemm_kernel<<<dim3(1, BS / 128), 256, 0, stream>>>(xb, wkt, bk, SCALE, 0, K, keysc, nullptr);
  gemm_kernel<<<dim3(1, BS / 128), 256, 0, stream>>>(xb, wqt, bq, 1.f,   0, K, qbf,   nullptr);
  gemm_kernel<<<dim3(D / 128, BS / 128), 256, 0, stream>>>(xb, wvt, bv, 1.f, 0, D, vbf, nullptr);

  attn_kernel<<<dim3(D / 128, S / 128, B), 256, 0, stream>>>(keysc, qbf, vbf, biasb, tmp);

  ln1_kernel<<<BS, 256, 0, stream>>>(x, tmp, g1, be1, resb);

  gemm_kernel<<<dim3(D / 128, BS / 128), 256, 0, stream>>>(resb, w1t, b1, 1.f, 1, D, nullptr, tmp);

  ln2_kernel<<<BS, 256, 0, stream>>>(resb, tmp, outp);
}
